// Spatial_Attn_78821239816811
// MI455X (gfx1250) — compile-verified, matrix operand order checked by reading, not run on hardware
//
#include <hip/hip_runtime.h>
#include <hip/hip_bf16.h>

typedef __attribute__((ext_vector_type(16))) _Float16 v16h;
typedef __attribute__((ext_vector_type(8)))  _Float16 v8h;
typedef __attribute__((ext_vector_type(8)))  float    v8f;

#define B_    512
#define T_    48
#define D_    128
#define F_    16
#define H_    512
#define K_IH  2048   // D*F
#define K_TOT 2560   // K_IH + H  (A = [x_hat | h])
#define N_G   2048   // 4*H gate columns

union Frag { v16h v; v8h h[2]; };

__device__ __forceinline__ void load_frag(const _Float16* __restrict__ p, int k0, Frag& f) {
  f.h[0] = *(const v8h*)(p + k0);        // K = khalf + 0..7
  f.h[1] = *(const v8h*)(p + k0 + 16);   // K = khalf + 16..23
}

// ---------------------------------------------------------------------------
// Setup: pack [w_ih | w_hh] to f16 (B-operand, row n, contiguous K), combine
// biases, zero c, zero the h-tail of the f16 A buffer (h0 = 0).
// ---------------------------------------------------------------------------
__global__ void k_setup(const float* __restrict__ w_ih, const float* __restrict__ w_hh,
                        const float* __restrict__ b_ih, const float* __restrict__ b_hh,
                        _Float16* __restrict__ wcomb, float* __restrict__ bias,
                        float* __restrict__ cbuf, _Float16* __restrict__ abuf) {
  long idx = (long)blockIdx.x * blockDim.x + threadIdx.x;
  const long W1 = (long)N_G * K_TOT;
  if (idx < W1) {
    int n = (int)(idx / K_TOT), k = (int)(idx % K_TOT);
    float v = (k < K_IH) ? w_ih[(long)n * K_IH + k] : w_hh[(long)n * H_ + (k - K_IH)];
    wcomb[idx] = (_Float16)v;
    return;
  }
  idx -= W1;
  if (idx < N_G) { bias[idx] = b_ih[idx] + b_hh[idx]; return; }
  idx -= N_G;
  if (idx < (long)B_ * H_) { cbuf[idx] = 0.0f; return; }
  idx -= (long)B_ * H_;
  if (idx < (long)B_ * H_) {
    int b = (int)(idx >> 9), n = (int)(idx & 511);
    abuf[(long)b * K_TOT + K_IH + n] = (_Float16)0.0f;
  }
}

// ---------------------------------------------------------------------------
// e_x[b,d] = sum_{t,f} input[b,t,d,f] * w_x[f*T + t] + attn_b
// One wave per (b,d); 32 lanes cover contiguous (t,f) chunks -> coalesced.
// ---------------------------------------------------------------------------
__global__ void k_ex(const float* __restrict__ input, const float* __restrict__ attn_w,
                     const float* __restrict__ attn_b, float* __restrict__ ex) {
  int wave = threadIdx.x >> 5, lane = threadIdx.x & 31;
  int pair = blockIdx.x * 8 + wave;            // B*D pairs
  int b = pair >> 7, d = pair & 127;
  const float* wx  = attn_w + 2 * H_;          // attn_w[2H:] is w_x (F*T)
  const float* src = input + (long)b * T_ * D_ * F_ + (long)d * F_;
  float s = 0.0f;
  for (int i = lane; i < T_ * F_; i += 32) {   // i = t*16 + f
    int t = i >> 4, f = i & 15;
    s += src[(long)t * D_ * F_ + f] * wx[f * T_ + t];
  }
  for (int off = 16; off > 0; off >>= 1) s += __shfl_down(s, off, 32);
  if (lane == 0) ex[pair] = s + attn_b[0];
}

// ---------------------------------------------------------------------------
// a[b,:] = softmax(e_x[b,:])  — h/c attention terms are per-row constants,
// shift-invariant under softmax, so attention is loop-invariant.
// ---------------------------------------------------------------------------
__global__ void k_softmax(const float* __restrict__ ex, float* __restrict__ a) {
  __shared__ float sm[D_];
  int d = threadIdx.x, b = blockIdx.x;
  float e = ex[b * D_ + d];
  sm[d] = e; __syncthreads();
  for (int s = D_ / 2; s > 0; s >>= 1) { if (d < s) sm[d] = fmaxf(sm[d], sm[d + s]); __syncthreads(); }
  float m = sm[0]; __syncthreads();
  float p = __expf(e - m);
  sm[d] = p; __syncthreads();
  for (int s = D_ / 2; s > 0; s >>= 1) { if (d < s) sm[d] += sm[d + s]; __syncthreads(); }
  a[b * D_ + d] = p / sm[0];
}

// ---------------------------------------------------------------------------
// x_hat[b,k] = a[b, k>>4] * input[b,t,:,:]  ->  f16 into A buffer cols [0,2048)
// ---------------------------------------------------------------------------
__global__ void k_xhat(const float* __restrict__ input, const float* __restrict__ a,
                       _Float16* __restrict__ abuf, int t) {
  int idx = blockIdx.x * blockDim.x + threadIdx.x;   // B * 2048
  int b = idx >> 11, k = idx & 2047, d = k >> 4;
  float v = input[(long)b * T_ * D_ * F_ + (long)t * D_ * F_ + k] * a[b * D_ + d];
  abuf[(long)b * K_TOT + k] = (_Float16)v;
}

// ---------------------------------------------------------------------------
// gates = A(512x2560,f16) x Wcomb^T(2560x2048,f16), f32 accumulation.
// Wave tile: 32(M) x 64(N) = 8 WMMA / K-step; register double-buffered
// (prefetch K+32 fragments before issuing WMMAs on K) to hide L2 latency.
// WG: 8 waves (4M x 2N) -> 128x128 tile. Grid: (2048/128, 512/128).
// ---------------------------------------------------------------------------
__global__ __launch_bounds__(256) void k_gemm(const _Float16* __restrict__ abuf,
                                              const _Float16* __restrict__ wcomb,
                                              float* __restrict__ gates) {
  const int wave = threadIdx.x >> 5, lane = threadIdx.x & 31;
  const int r = lane & 15, lh = lane >> 4;           // row/col in tile, K-half select
  const int waveM = wave & 3, waveN = wave >> 2;
  const int m0 = blockIdx.y * 128 + waveM * 32;
  const int n0 = blockIdx.x * 128 + waveN * 64;

  const _Float16* Ap[2];
  const _Float16* Bp[4];
#pragma unroll
  for (int mi = 0; mi < 2; ++mi)
    Ap[mi] = abuf + (long)(m0 + mi * 16 + r) * K_TOT + lh * 8;
#pragma unroll
  for (int j = 0; j < 4; ++j)
    Bp[j] = wcomb + (long)(n0 + j * 16 + r) * K_TOT + lh * 8;

  v8f acc[2][4] = {};
  Frag a0[2], b0[4], a1[2], b1[4];

  // prologue: slab 0 into buffer 0
#pragma unroll
  for (int mi = 0; mi < 2; ++mi) load_frag(Ap[mi], 0, a0[mi]);
#pragma unroll
  for (int j = 0; j < 4; ++j)    load_frag(Bp[j], 0, b0[j]);

  for (int k0 = 0; k0 < K_TOT; k0 += 64) {
    // prefetch slab k0+32 into buffer 1 (always in range: last k0 = 2496)
#pragma unroll
    for (int mi = 0; mi < 2; ++mi) load_frag(Ap[mi], k0 + 32, a1[mi]);
#pragma unroll
    for (int j = 0; j < 4; ++j)    load_frag(Bp[j], k0 + 32, b1[j]);
    // compute on buffer 0 (slab k0)
#pragma unroll
    for (int mi = 0; mi < 2; ++mi)
#pragma unroll
      for (int j = 0; j < 4; ++j)
        acc[mi][j] = __builtin_amdgcn_wmma_f32_16x16x32_f16(
            false, a0[mi].v, false, b0[j].v, (short)0, acc[mi][j], false, false);

    if (k0 + 64 < K_TOT) {
      // prefetch slab k0+64 into buffer 0
#pragma unroll
      for (int mi = 0; mi < 2; ++mi) load_frag(Ap[mi], k0 + 64, a0[mi]);
#pragma unroll
      for (int j = 0; j < 4; ++j)    load_frag(Bp[j], k0 + 64, b0[j]);
    }
    // compute on buffer 1 (slab k0+32)
#pragma unroll
    for (int mi = 0; mi < 2; ++mi)
#pragma unroll
      for (int j = 0; j < 4; ++j)
        acc[mi][j] = __builtin_amdgcn_wmma_f32_16x16x32_f16(
            false, a1[mi].v, false, b1[j].v, (short)0, acc[mi][j], false, false);
  }

#pragma unroll
  for (int mi = 0; mi < 2; ++mi)
#pragma unroll
    for (int j = 0; j < 4; ++j)
#pragma unroll
      for (int v = 0; v < 8; ++v) {                  // D layout: M = v + 8*lh, N = r
        int m = m0 + mi * 16 + lh * 8 + v;
        int n = n0 + j * 16 + r;
        gates[(long)m * N_G + n] = acc[mi][j][v];
      }
}

// ---------------------------------------------------------------------------
// LSTM pointwise epilogue: update c, emit h (f32 to out, f16 into A tail).
// ---------------------------------------------------------------------------
__device__ __forceinline__ float sigm(float x) { return 1.0f / (1.0f + __expf(-x)); }

__global__ void k_lstm(const float* __restrict__ gates, const float* __restrict__ bias,
                       float* __restrict__ cbuf, _Float16* __restrict__ abuf,
                       float* __restrict__ out, int t) {
  int idx = blockIdx.x * blockDim.x + threadIdx.x;   // B * H
  int b = idx >> 9, n = idx & 511;
  const float* g = gates + (long)b * N_G;
  float gi = g[n]        + bias[n];
  float gf = g[n +  512] + bias[n +  512];
  float gg = g[n + 1024] + bias[n + 1024];
  float go = g[n + 1536] + bias[n + 1536];
  float c_old = cbuf[idx];
  float c_new = sigm(gf) * c_old + sigm(gi) * tanhf(gg);
  float h_new = sigm(go) * tanhf(c_new);
  cbuf[idx] = c_new;
  out[((long)b * T_ + t) * H_ + n] = h_new;
  abuf[(long)b * K_TOT + K_IH + n] = (_Float16)h_new;
}

// ---------------------------------------------------------------------------
extern "C" void kernel_launch(void* const* d_in, const int* in_sizes, int n_in,
                              void* d_out, int out_size, void* d_ws, size_t ws_size,
                              hipStream_t stream) {
  (void)in_sizes; (void)n_in; (void)out_size; (void)ws_size;
  const float* input  = (const float*)d_in[0];
  const float* attn_w = (const float*)d_in[1];
  const float* attn_b = (const float*)d_in[2];
  const float* w_ih   = (const float*)d_in[3];
  const float* w_hh   = (const float*)d_in[4];
  const float* b_ih   = (const float*)d_in[5];
  const float* b_hh   = (const float*)d_in[6];
  float* out = (float*)d_out;

  char* ws = (char*)d_ws;
  size_t off = 0;
  _Float16* wcomb = (_Float16*)(ws + off); off += (size_t)N_G * K_TOT * 2;  // 10.5 MB
  float*    bias  = (float*)(ws + off);    off += (size_t)N_G * 4;          //  8 KB
  float*    exb   = (float*)(ws + off);    off += (size_t)B_ * D_ * 4;      // 256 KB
  float*    ab    = (float*)(ws + off);    off += (size_t)B_ * D_ * 4;      // 256 KB
  float*    cbuf  = (float*)(ws + off);    off += (size_t)B_ * H_ * 4;      //  1 MB
  _Float16* abuf  = (_Float16*)(ws + off); off += (size_t)B_ * K_TOT * 2;   // 2.6 MB
  float*    gates = (float*)(ws + off);    off += (size_t)B_ * N_G * 4;     //  4 MB

  long Wsetup = (long)N_G * K_TOT + N_G + (long)B_ * H_ * 2;
  k_setup<<<(int)((Wsetup + 255) / 256), 256, 0, stream>>>(w_ih, w_hh, b_ih, b_hh,
                                                           wcomb, bias, cbuf, abuf);
  k_ex<<<(B_ * D_) / 8, 256, 0, stream>>>(input, attn_w, attn_b, exb);
  k_softmax<<<B_, D_, 0, stream>>>(exb, ab);

  dim3 ggrid(N_G / 128, B_ / 128);
  for (int t = 0; t < T_; ++t) {
    k_xhat<<<(B_ * K_IH) / 256, 256, 0, stream>>>(input, ab, abuf, t);
    k_gemm<<<ggrid, 256, 0, stream>>>(abuf, wcomb, gates);
    k_lstm<<<(B_ * H_) / 256, 256, 0, stream>>>(gates, bias, cbuf, abuf, out, t);
  }
}
